// ExpertsFeedForward_2284922602057
// MI455X (gfx1250) — matrix operand order repaired and checked by reading, not compiled, not run on hardware
//
#include <hip/hip_runtime.h>
#include <hip/hip_bf16.h>
#include <math.h>

// ---------------- problem constants (from reference) ----------------
#define DD    1024          // model dim
#define HH    4096          // expert hidden dim
#define EE    8             // experts
#define NTOK  16384         // B*S
#define CAP   3072          // int(1.5*N/E)

// ---------------- FFN tiling ----------------
#define BM    32            // token rows per workgroup
#define HC    128           // H-chunk per iteration (8 waves * 16 cols)
#define KC    128           // K-chunk staged per step
#define XSTR  1040          // bf16 row stride for x tile (16B aligned, bank-friendly)
#define BSTR  136           // bf16 column stride for staged B (272B: 16B aligned, conflict-free)
#define HSTR  136           // bf16 row stride for h tile (16B aligned)

typedef __attribute__((ext_vector_type(16))) __bf16 v16bf;
typedef __attribute__((ext_vector_type(8)))  __bf16 v8bf;
typedef __attribute__((ext_vector_type(4)))  __bf16 v4bf;
typedef __attribute__((ext_vector_type(8)))  float  v8f;
typedef __attribute__((ext_vector_type(4)))  unsigned int v4u;
typedef __attribute__((ext_vector_type(8)))  int    v8i_t;
typedef __attribute__((ext_vector_type(4)))  int    v4i_t;

__device__ __forceinline__ v8f zero8() {
  v8f z;
#pragma unroll
  for (int i = 0; i < 8; ++i) z[i] = 0.0f;
  return z;
}

// A fragment (16x32 bf16, M x K), row-major source with given stride.
__device__ __forceinline__ v16bf load_a_frag(const __bf16* base, int stride,
                                             int kbase, int lM, int half) {
  const __bf16* p = base + lM * stride + kbase + 8 * half;
  v8bf lo = *(const v8bf*)p;
  v8bf hi = *(const v8bf*)(p + 16);
  v16bf a;
#pragma unroll
  for (int i = 0; i < 8; ++i) { a[i] = lo[i]; a[i + 8] = hi[i]; }
  return a;
}

// B fragment (32x16 bf16, K x N), column-major source (K contiguous per column).
__device__ __forceinline__ v16bf load_b_frag(const __bf16* colbase, int kbase, int half) {
  const __bf16* p = colbase + kbase + 8 * half;
  v8bf lo = *(const v8bf*)p;
  v8bf hi = *(const v8bf*)(p + 16);
  v16bf b;
#pragma unroll
  for (int i = 0; i < 8; ++i) { b[i] = lo[i]; b[i + 8] = hi[i]; }
  return b;
}

__device__ __forceinline__ v8f wmma_bf16(v16bf a, v16bf b, v8f c) {
  return __builtin_amdgcn_wmma_f32_16x16x32_bf16(false, a, false, b,
                                                 (short)0, c, false, false);
}

__device__ __forceinline__ v4bf pack4(float a, float b, float c, float d) {
  v4bf r;
  r[0] = (__bf16)a; r[1] = (__bf16)b; r[2] = (__bf16)c; r[3] = (__bf16)d;
  return r;
}

// LDS byte offset of a __shared__ object (generic -> addrspace(3) -> int)
__device__ __forceinline__ unsigned lds_off(const void* p) {
  return (unsigned)(unsigned long long)(__attribute__((address_space(3))) const void*)p;
}

// ---- Tensor Data Mover: 2D tile (128 rows x 128 bf16) -> LDS with 4-dword row pad ----
// D# packing per CDNA5 ISA. Row in LDS = 256B data + 16B pad = BSTR*2.
__device__ __forceinline__ void tdm_load_tile(unsigned lds_addr, const __bf16* gptr,
                                              unsigned stride_elems) {
  unsigned long long ga = (unsigned long long)(size_t)gptr;
  v4u g0;
  g0[0] = 1u;                                             // count=1
  g0[1] = lds_addr;                                       // lds_addr (bytes)
  g0[2] = (unsigned)(ga & 0xffffffffu);                   // global_addr lo
  g0[3] = (unsigned)((ga >> 32) & 0x01ffffffu) | (2u << 30);  // addr hi | type=2
  v8i_t g1;
  g1[0] = (int)((1u << 16) | (1u << 20) | (5u << 22) | (3u << 25));
  //        data_size=2B    pad_enable   pad_int=64dw   pad_amt=4dw
  g1[1] = (int)((stride_elems & 0xffffu) << 16);          // tensor_dim0 lo16 @[31:16]
  g1[2] = (int)(((stride_elems >> 16) & 0xffffu) | (128u << 16)); // dim0 hi | tensor_dim1 lo16
  g1[3] = (int)(128u << 16);                              // tensor_dim1 hi=0 | tile_dim0=128
  g1[4] = (int)128;                                       // tile_dim1=128, tile_dim2=0
  g1[5] = (int)stride_elems;                              // tensor_dim0_stride lo32
  g1[6] = 0;                                              // stride hi | dim1_stride lo
  g1[7] = 0;
  v4i_t z; z[0] = z[1] = z[2] = z[3] = 0;
#if __clang_major__ >= 23
  v8i_t z8; 
#pragma unroll
  for (int i = 0; i < 8; ++i) z8[i] = 0;
  __builtin_amdgcn_tensor_load_to_lds(g0, g1, z, z, z8, 0);
#else
  __builtin_amdgcn_tensor_load_to_lds(g0, g1, z, z, 0);
#endif
}

// ---------------- init: importance=0, tok_idx=pad ----------------
__global__ void init_kernel(float* __restrict__ importance, int* __restrict__ tokidx) {
  int i = blockIdx.x * blockDim.x + threadIdx.x;
  if (i < EE) importance[i] = 0.0f;
  if (i < EE * CAP) tokidx[i] = NTOK;
}

// ---------------- router: one wave per token ----------------
__global__ __launch_bounds__(256) void router_kernel(
    const float* __restrict__ x, const float* __restrict__ Wr,
    const float* __restrict__ br, int* __restrict__ gidx,
    float* __restrict__ gval, float* __restrict__ importance) {
  __shared__ float imp_s[EE];
  int tid = threadIdx.x;
  int w = tid >> 5, lane = tid & 31;
  if (tid < EE) imp_s[tid] = 0.0f;
  __syncthreads();

  int t = blockIdx.x * 8 + w;
  const float* xr = x + (size_t)t * DD;
  float acc[EE];
#pragma unroll
  for (int e = 0; e < EE; ++e) acc[e] = 0.0f;

  for (int d = lane * 4; d < DD; d += 128) {
    float4 xv = *(const float4*)(xr + d);
    const float* wr = Wr + (size_t)d * EE;
    float xs4[4] = {xv.x, xv.y, xv.z, xv.w};
#pragma unroll
    for (int q = 0; q < 4; ++q)
#pragma unroll
      for (int e = 0; e < EE; ++e) acc[e] += xs4[q] * wr[q * EE + e];
  }
#pragma unroll
  for (int off = 16; off > 0; off >>= 1)
#pragma unroll
    for (int e = 0; e < EE; ++e) acc[e] += __shfl_xor(acc[e], off, 32);

  if (lane == 0) {
    float lg[EE];
    float mx = -3.0e38f; int ai = 0;
#pragma unroll
    for (int e = 0; e < EE; ++e) {
      lg[e] = acc[e] + br[e];
      if (lg[e] > mx) { mx = lg[e]; ai = e; }
    }
    float s = 0.0f, p[EE];
#pragma unroll
    for (int e = 0; e < EE; ++e) { p[e] = __expf(lg[e] - mx); s += p[e]; }
    float inv = 1.0f / s;
    gval[t] = p[ai] * inv;
    gidx[t] = ai;
#pragma unroll
    for (int e = 0; e < EE; ++e) atomicAdd(&imp_s[e], p[e] * inv);
  }
  __syncthreads();
  if (tid < EE) atomicAdd(&importance[tid], imp_s[tid]);
}

// ---------------- dispatch: FIFO capacity assignment (single block) ----------------
__global__ __launch_bounds__(256) void dispatch_kernel(
    const int* __restrict__ gidx, int* __restrict__ keep, int* __restrict__ tokidx) {
  __shared__ int running[EE];
  __shared__ int wavecnt[8][EE];
  int tid = threadIdx.x, lane = tid & 31, w = tid >> 5;
  if (tid < EE) running[tid] = 0;
  __syncthreads();

  for (int base = 0; base < NTOK; base += 256) {
    int t = base + tid;
    int e = gidx[t];
    unsigned mymask = 0;
#pragma unroll
    for (int q = 0; q < EE; ++q) {
      unsigned m = __builtin_amdgcn_ballot_w32(e == q);
      if (e == q) mymask = m;
      if (lane == q) wavecnt[w][q] = __popc(m);
    }
    __syncthreads();
    int rank = __popc(mymask & ((1u << lane) - 1u));
    int pre = running[e];
    for (int ww = 0; ww < w; ++ww) pre += wavecnt[ww][e];
    int pos = pre + rank;
    int kp = (pos < CAP) ? 1 : 0;
    keep[t] = kp;
    if (kp) tokidx[e * CAP + pos] = t;
    __syncthreads();
    if (tid < EE) {
      int s = 0;
#pragma unroll
      for (int ww = 0; ww < 8; ++ww) s += wavecnt[ww][tid];
      running[tid] += s;
    }
    __syncthreads();
  }
}

// ---------------- weight convert+transpose: src[R][C] fp32 -> dst[C][R] bf16 ----------------
// blockIdx.y = expert; 64x64 tiles through LDS, both sides coalesced.
__global__ __launch_bounds__(256) void convert_transpose_kernel(
    const float* __restrict__ src, unsigned short* __restrict__ dst_us, int R, int C) {
  __shared__ float tile[64][65];
  __bf16* dst = (__bf16*)dst_us;
  const float* s = src + (size_t)blockIdx.y * R * C;
  __bf16* d = dst + (size_t)blockIdx.y * R * C;
  int tpr = C >> 6;
  int tr = blockIdx.x / tpr, tc = blockIdx.x % tpr;
  int r0 = tr << 6, c0 = tc << 6;
  int j = threadIdx.x & 63, i0 = threadIdx.x >> 6;
#pragma unroll
  for (int i = i0; i < 64; i += 4)
    tile[i][j] = s[(size_t)(r0 + i) * C + c0 + j];
  __syncthreads();
  int r = threadIdx.x & 63, cc0 = threadIdx.x >> 6;
#pragma unroll
  for (int c = cc0; c < 64; c += 4)
    d[(size_t)(c0 + c) * R + r0 + r] = (__bf16)tile[r][c];
}

// ================= shared FFN body pieces =================
struct FfnCtx {
  int e, rb, tid, lane, w, lN, half;
};

__device__ __forceinline__ void ffn_prologue(
    const FfnCtx& c, const float* x, const int* tokidx, const float* gate,
    int* toks, float* gates, __bf16* xs) {
  if (c.tid < BM) {
    int t = tokidx[c.e * CAP + c.rb * BM + c.tid];
    toks[c.tid] = t;
    gates[c.tid] = (t < NTOK) ? gate[t] : 0.0f;
  }
  __syncthreads();
#pragma unroll 4
  for (int i = 0; i < (BM * DD / 4) / 256; ++i) {
    int q = c.tid + 256 * i;
    int r = q >> 8, cc = (q & 255) * 4;
    int t = toks[r];
    float4 v = (t < NTOK) ? *(const float4*)(x + (size_t)t * DD + cc)
                          : make_float4(0.f, 0.f, 0.f, 0.f);
    *(v4bf*)(&xs[r * XSTR + cc]) = pack4(v.x, v.y, v.z, v.w);
  }
  __syncthreads();
}

__device__ __forceinline__ void ffn_epilogue(
    const FfnCtx& c, const float* b2e, const int* toks, const float* gates,
    const v8f* acc2a, const v8f* acc2b, float* out) {
#pragma unroll
  for (int s = 0; s < 8; ++s) {
    int ncol = s * 128 + 16 * c.w + c.lN;
    float bias2 = b2e[ncol];
#pragma unroll
    for (int v = 0; v < 8; ++v) {
      int M0 = v + 8 * c.half;
      int t0 = toks[M0];
      if (t0 < NTOK)
        out[(size_t)t0 * DD + ncol] = (acc2a[s][v] + bias2) * gates[M0];
      int t1 = toks[16 + M0];
      if (t1 < NTOK)
        out[(size_t)t1 * DD + ncol] = (acc2b[s][v] + bias2) * gates[16 + M0];
    }
  }
}

// ---------------- FFN, TDM path: bf16 transposed weights streamed by the Tensor Data Mover ----
__global__ __launch_bounds__(256) void moe_ffn_tdm_kernel(
    const float* __restrict__ x, const unsigned short* __restrict__ W1bT_us,
    const float* __restrict__ b1, const unsigned short* __restrict__ W2bT_us,
    const float* __restrict__ b2, const int* __restrict__ tokidx,
    const float* __restrict__ gate, float* __restrict__ out) {
  __shared__ __bf16 xs[BM * XSTR];      // ~66.6 KB
  __shared__ __bf16 bufA[HC * BSTR];    // ~34.8 KB (TDM dest, ping)
  __shared__ __bf16 bufB[HC * BSTR];    // ~34.8 KB (TDM dest, pong)
  __shared__ __bf16 hs[BM * HSTR];      // ~8.7 KB
  __shared__ int   toks[BM];
  __shared__ float gates[BM];

  FfnCtx c;
  c.e = blockIdx.x / (CAP / BM); c.rb = blockIdx.x % (CAP / BM);
  c.tid = threadIdx.x; c.lane = c.tid & 31; c.w = c.tid >> 5;
  c.lN = c.lane & 15; c.half = c.lane >> 4;

  ffn_prologue(c, x, tokidx, gate, toks, gates, xs);

  const __bf16* W1e = (const __bf16*)W1bT_us + (size_t)c.e * DD * HH; // [H][D] layout
  const __bf16* W2e = (const __bf16*)W2bT_us + (size_t)c.e * HH * DD; // [D][H] layout
  const float* b1e = b1 + (size_t)c.e * HH;
  const float* b2e = b2 + (size_t)c.e * DD;
  const unsigned offA = lds_off(bufA), offB = lds_off(bufB);

  v8f acc2a[8], acc2b[8];
#pragma unroll
  for (int s = 0; s < 8; ++s) { acc2a[s] = zero8(); acc2b[s] = zero8(); }

  for (int h0 = 0; h0 < HH; h0 += HC) {
    float bias1 = b1e[h0 + 16 * c.w + c.lN];
    v8f acc1a = zero8(), acc1b = zero8();

    // GEMM1 B-tiles: W1bT rows h0..h0+127, cols kc..kc+127 (row stride DD)
    if (c.w == 0) {
      tdm_load_tile(offA, W1e + (size_t)h0 * DD + 0 * KC, DD);
      tdm_load_tile(offB, W1e + (size_t)h0 * DD + 1 * KC, DD);
    }
#pragma unroll 1
    for (int ci = 0; ci < DD / KC; ++ci) {   // 8 chunks
      if (ci < DD / KC - 1) __builtin_amdgcn_s_wait_tensorcnt(1);
      else                  __builtin_amdgcn_s_wait_tensorcnt(0);
      __syncthreads();
      const __bf16* bb_base = (ci & 1) ? bufB : bufA;
#pragma unroll
      for (int ks = 0; ks < KC; ks += 32) {
        v16bf bb = load_b_frag(bb_base + (16 * c.w + c.lN) * BSTR, ks, c.half);
        v16bf a0 = load_a_frag(xs, XSTR, ci * KC + ks, c.lN, c.half);
        v16bf a1 = load_a_frag(xs + 16 * XSTR, XSTR, ci * KC + ks, c.lN, c.half);
        acc1a = wmma_bf16(a0, bb, acc1a);
        acc1b = wmma_bf16(a1, bb, acc1b);
      }
      __syncthreads();
      if (c.w == 0 && ci + 2 < DD / KC)
        tdm_load_tile((ci & 1) ? offB : offA, W1e + (size_t)h0 * DD + (ci + 2) * KC, DD);
    }
    // bias + silu -> bf16 h tile
#pragma unroll
    for (int v = 0; v < 8; ++v) {
      float hv = acc1a[v] + bias1;
      hv = hv * __builtin_amdgcn_rcpf(1.0f + __expf(-hv));
      hs[(v + 8 * c.half) * HSTR + 16 * c.w + c.lN] = (__bf16)hv;
      float hw = acc1b[v] + bias1;
      hw = hw * __builtin_amdgcn_rcpf(1.0f + __expf(-hw));
      hs[(16 + v + 8 * c.half) * HSTR + 16 * c.w + c.lN] = (__bf16)hw;
    }
    __syncthreads();

    // GEMM2 B-tiles: W2bT rows s*128..+127 (out col), cols h0..h0+127 (row stride HH)
    if (c.w == 0) {
      tdm_load_tile(offA, W2e + (size_t)(0 * 128) * HH + h0, HH);
      tdm_load_tile(offB, W2e + (size_t)(1 * 128) * HH + h0, HH);
    }
#pragma unroll 1
    for (int s = 0; s < 8; ++s) {
      if (s < 7) __builtin_amdgcn_s_wait_tensorcnt(1);
      else       __builtin_amdgcn_s_wait_tensorcnt(0);
      __syncthreads();
      const __bf16* bb_base = (s & 1) ? bufB : bufA;
#pragma unroll
      for (int ks = 0; ks < KC; ks += 32) {
        v16bf bb = load_b_frag(bb_base + (16 * c.w + c.lN) * BSTR, ks, c.half);
        v16bf a0 = load_a_frag(hs, HSTR, ks, c.lN, c.half);
        v16bf a1 = load_a_frag(hs + 16 * HSTR, HSTR, ks, c.lN, c.half);
        acc2a[s] = wmma_bf16(a0, bb, acc2a[s]);
        acc2b[s] = wmma_bf16(a1, bb, acc2b[s]);
      }
      __syncthreads();
      if (c.w == 0 && s + 2 < 8)
        tdm_load_tile((s & 1) ? offB : offA, W2e + (size_t)((s + 2) * 128) * HH + h0, HH);
    }
  }
  ffn_epilogue(c, b2e, toks, gates, acc2a, acc2b, out);
}

// ---------------- FFN, fallback path: VALU-staged fp32 weights (small ws_size) ----------------
__global__ __launch_bounds__(256) void moe_ffn_kernel(
    const float* __restrict__ x, const float* __restrict__ W1,
    const float* __restrict__ b1, const float* __restrict__ W2,
    const float* __restrict__ b2, const int* __restrict__ tokidx,
    const float* __restrict__ gate, float* __restrict__ out) {
  __shared__ __bf16 xs[BM * XSTR];
  __shared__ __bf16 buf[HC * BSTR];
  __shared__ __bf16 hs[BM * HSTR];
  __shared__ int   toks[BM];
  __shared__ float gates[BM];

  FfnCtx c;
  c.e = blockIdx.x / (CAP / BM); c.rb = blockIdx.x % (CAP / BM);
  c.tid = threadIdx.x; c.lane = c.tid & 31; c.w = c.tid >> 5;
  c.lN = c.lane & 15; c.half = c.lane >> 4;

  ffn_prologue(c, x, tokidx, gate, toks, gates, xs);

  const float* W1e = W1 + (size_t)c.e * DD * HH;
  const float* W2e = W2 + (size_t)c.e * HH * DD;
  const float* b1e = b1 + (size_t)c.e * HH;
  const float* b2e = b2 + (size_t)c.e * DD;

  v8f acc2a[8], acc2b[8];
#pragma unroll
  for (int s = 0; s < 8; ++s) { acc2a[s] = zero8(); acc2b[s] = zero8(); }

  for (int h0 = 0; h0 < HH; h0 += HC) {
    float bias1 = b1e[h0 + 16 * c.w + c.lN];
    v8f acc1a = zero8(), acc1b = zero8();
#pragma unroll 1
    for (int kc = 0; kc < DD; kc += KC) {
#pragma unroll 4
      for (int i = 0; i < (KC * HC / 4) / 256; ++i) {
        int q = c.tid + 256 * i;
        int n = q & 127, k0 = (q >> 7) * 4;
        const float* g = W1e + (size_t)(kc + k0) * HH + h0 + n;
        *(v4bf*)(&buf[n * BSTR + k0]) = pack4(g[0], g[HH], g[2 * HH], g[3 * HH]);
      }
      if (kc + KC < DD)
        __builtin_prefetch(&W1e[(size_t)(kc + KC + (c.tid >> 7) * 4) * HH + h0 + (c.tid & 127)], 0, 1);
      __syncthreads();
#pragma unroll
      for (int ks = 0; ks < KC; ks += 32) {
        v16bf bb = load_b_frag(buf + (16 * c.w + c.lN) * BSTR, ks, c.half);
        v16bf a0 = load_a_frag(xs, XSTR, kc + ks, c.lN, c.half);
        v16bf a1 = load_a_frag(xs + 16 * XSTR, XSTR, kc + ks, c.lN, c.half);
        acc1a = wmma_bf16(a0, bb, acc1a);
        acc1b = wmma_bf16(a1, bb, acc1b);
      }
      __syncthreads();
    }
#pragma unroll
    for (int v = 0; v < 8; ++v) {
      float hv = acc1a[v] + bias1;
      hv = hv * __builtin_amdgcn_rcpf(1.0f + __expf(-hv));
      hs[(v + 8 * c.half) * HSTR + 16 * c.w + c.lN] = (__bf16)hv;
      float hw = acc1b[v] + bias1;
      hw = hw * __builtin_amdgcn_rcpf(1.0f + __expf(-hw));
      hs[(16 + v + 8 * c.half) * HSTR + 16 * c.w + c.lN] = (__bf16)hw;
    }
    __syncthreads();
#pragma unroll 1
    for (int s = 0; s < 8; ++s) {
#pragma unroll 4
      for (int i = 0; i < (KC * 128 / 4) / 256; ++i) {
        int q = c.tid + 256 * i;
        int n = q & 127, k0 = (q >> 7) * 4;
        const float* g = W2e + (size_t)(h0 + k0) * DD + s * 128 + n;
        *(v4bf*)(&buf[n * BSTR + k0]) = pack4(g[0], g[DD], g[2 * DD], g[3 * DD]);
      }
      if (s + 1 < 8)
        __builtin_prefetch(&W2e[(size_t)(h0 + (c.tid >> 7) * 4) * DD + (s + 1) * 128 + (c.tid & 127)], 0, 1);
      __syncthreads();
#pragma unroll
      for (int ks = 0; ks < KC; ks += 32) {
        v16bf bb = load_b_frag(buf + (16 * c.w + c.lN) * BSTR, ks, c.half);
        v16bf a0 = load_a_frag(hs, HSTR, ks, c.lN, c.half);
        v16bf a1 = load_a_frag(hs + 16 * HSTR, HSTR, ks, c.lN, c.half);
        acc2a[s] = wmma_bf16(a0, bb, acc2a[s]);
        acc2b[s] = wmma_bf16(a1, bb, acc2b[s]);
      }
      __syncthreads();
    }
  }
  ffn_epilogue(c, b2e, toks, gates, acc2a, acc2b, out);
}

// ---------------- dropped tokens pass through: out = x * gate ----------------
__global__ __launch_bounds__(256) void passthrough_kernel(
    const float* __restrict__ x, const int* __restrict__ keep,
    const float* __restrict__ gate, float* __restrict__ out) {
  int t = blockIdx.x;
  if (keep[t]) return;
  float g = gate[t];
  for (int c = threadIdx.x * 4; c < DD; c += 1024) {
    float4 v = *(const float4*)(x + (size_t)t * DD + c);
    v.x *= g; v.y *= g; v.z *= g; v.w *= g;
    *(float4*)(out + (size_t)t * DD + c) = v;
  }
}

// ---------------- aux losses ----------------
__global__ void finalize_kernel(const float* __restrict__ importance, float* __restrict__ out) {
  if (threadIdx.x == 0 && blockIdx.x == 0) {
    float s = 0.0f;
#pragma unroll
    for (int e = 0; e < EE; ++e) s += importance[e];
    float m = s / (float)EE;
    float var = 0.0f;
#pragma unroll
    for (int e = 0; e < EE; ++e) { float d = importance[e] - m; var += d * d; }
    var /= (float)EE;
    out[(size_t)NTOK * DD]     = 1.0f;
    out[(size_t)NTOK * DD + 1] = var / (m * m);
  }
}

extern "C" void kernel_launch(void* const* d_in, const int* in_sizes, int n_in,
                              void* d_out, int out_size, void* d_ws, size_t ws_size,
                              hipStream_t stream) {
  (void)in_sizes; (void)n_in; (void)out_size;
  const float* x  = (const float*)d_in[0];
  const float* Wr = (const float*)d_in[1];
  const float* br = (const float*)d_in[2];
  const float* W1 = (const float*)d_in[3];
  const float* b1 = (const float*)d_in[4];
  const float* W2 = (const float*)d_in[5];
  const float* b2 = (const float*)d_in[6];
  float* out = (float*)d_out;

  char* ws = (char*)d_ws;
  float* importance = (float*)ws;                 ws += 256;
  int*   gidx       = (int*)ws;                   ws += (size_t)NTOK * 4;
  float* gval       = (float*)ws;                 ws += (size_t)NTOK * 4;
  int*   keep       = (int*)ws;                   ws += (size_t)NTOK * 4;
  int*   tokidx     = (int*)ws;                   ws += (size_t)EE * CAP * 4;

  size_t ctrl = (size_t)(ws - (char*)d_ws);
  ctrl = (ctrl + 255) & ~(size_t)255;
  size_t wbytes = (size_t)EE * DD * HH * 2;       // one bf16 weight tensor
  bool use_tdm = ws_size >= ctrl + 2 * wbytes;
  unsigned short* W1bT = (unsigned short*)((char*)d_ws + ctrl);
  unsigned short* W2bT = W1bT + (size_t)EE * DD * HH;

  init_kernel<<<(EE * CAP + 255) / 256, 256, 0, stream>>>(importance, tokidx);
  router_kernel<<<NTOK / 8, 256, 0, stream>>>(x, Wr, br, gidx, gval, importance);
  dispatch_kernel<<<1, 256, 0, stream>>>(gidx, keep, tokidx);
  if (use_tdm) {
    // W1 [E][D][H] -> W1bT [E][H][D];  W2 [E][H][D] -> W2bT [E][D][H]
    convert_transpose_kernel<<<dim3((DD / 64) * (HH / 64), EE), 256, 0, stream>>>(W1, W1bT, DD, HH);
    convert_transpose_kernel<<<dim3((HH / 64) * (DD / 64), EE), 256, 0, stream>>>(W2, W2bT, HH, DD);
    moe_ffn_tdm_kernel<<<EE * (CAP / BM), 256, 0, stream>>>(x, W1bT, b1, W2bT, b2, tokidx, gval, out);
  } else {
    moe_ffn_kernel<<<EE * (CAP / BM), 256, 0, stream>>>(x, W1, b1, W2, b2, tokidx, gval, out);
  }
  passthrough_kernel<<<NTOK, 256, 0, stream>>>(x, keep, gval, out);
  finalize_kernel<<<1, 1, 0, stream>>>(importance, out);
}
